// YOLOLayer_8022998909599
// MI455X (gfx1250) — compile-verified, matrix operand order checked by reading, not run on hardware
//
#include <hip/hip_runtime.h>

typedef __attribute__((ext_vector_type(16))) _Float16 v16h;
typedef __attribute__((ext_vector_type(8)))  float    v8f;

#define FSIZE   76
#define NPIX    (FSIZE * FSIZE)   // 5776
#define NCHAN   256
#define NOUT    255               // 3 anchors * 85
#define NBATCH  32
#define MT_PER_B (NPIX / 16)      // 361 exact
#define NTILES  16                // ceil(255/16) with zero pad
#define KSTEPS  8                 // 256 / 32

// ---------------------------------------------------------------------------
// Prep: swizzle W [255,256] f32 -> f16 B-fragments in the exact wave32 WMMA
// B-matrix VGPR layout (mirrors the documented 16-bit A 16x32 layout with
// lane = N). Fragment (nt, ks): 32 lanes x 16 halves, contiguous per lane.
// o = 255 padded with zeros so the 16th N-tile is safe to multiply.
// ---------------------------------------------------------------------------
__global__ __launch_bounds__(256) void yolo_prep_b(const float* __restrict__ W,
                                                   _Float16* __restrict__ bf) {
  int id = blockIdx.x * 256 + threadIdx.x;          // 4096 fragments-lanes
  if (id >= NTILES * KSTEPS * 32) return;
  int lane = id & 31;
  int ks   = (id >> 5) & 7;
  int nt   = id >> 8;
  int lm = lane & 15, hi = lane >> 4;
  int o = nt * 16 + lm;                              // N index (out channel)
  _Float16* dst = bf + (size_t)id * 16;
#pragma unroll
  for (int t = 0; t < 16; ++t) {
    // 16-bit operand layout: half t of lane -> K = t + 8*hi + (t>=8 ? 8 : 0)
    int c = 32 * ks + t + 8 * hi + ((t >= 8) ? 8 : 0);
    float v = (o < NOUT) ? W[o * NCHAN + c] : 0.0f;
    dst[t] = (_Float16)v;
  }
}

// ---------------------------------------------------------------------------
// Main: one wave per 16-pixel tile. A (16px x 256ch) held in 8 v16h fragments,
// loaded once from HBM; B streamed from L2-resident swizzled workspace.
// 128 x v_wmma_f32_16x16x32_f16 per wave, fused YOLO decode epilogue.
// ---------------------------------------------------------------------------
__global__ __launch_bounds__(256) void yolo_main(const float* __restrict__ x,
                                                 const float* __restrict__ bias,
                                                 const _Float16* __restrict__ bfrag,
                                                 float* __restrict__ out) {
  const int lane  = threadIdx.x & 31;
  const int wave  = threadIdx.x >> 5;
  const int mtile = blockIdx.x * 8 + wave;           // 11552 tiles total
  const int b     = mtile / MT_PER_B;
  const int pt    = mtile - b * MT_PER_B;
  const int pbase = pt * 16;
  const int lm = lane & 15, hi = lane >> 4;

  // per-lane base: pixel = pbase + lm; channel stride NPIX floats.
  const float* xp = x + (size_t)b * NCHAN * NPIX + pbase + lm;

  // ---- load + convert A fragments (each x dword touched exactly once) ----
  v16h A[KSTEPS];
#pragma unroll
  for (int ks = 0; ks < KSTEPS; ++ks) {
#pragma unroll
    for (int t = 0; t < 16; ++t) {
      int c = 32 * ks + t + 8 * hi + ((t >= 8) ? 8 : 0);
      A[ks][t] = (_Float16)xp[(size_t)c * NPIX];     // imm offset < 8 MB
    }
  }

  // pixel -> (row, col) per C-matrix row r (M = r + 8*hi), hoisted
  int prow[8], pcol[8];
#pragma unroll
  for (int r = 0; r < 8; ++r) {
    int p = pbase + r + 8 * hi;
    prow[r] = p / FSIZE;
    pcol[r] = p - prow[r] * FSIZE;
  }

  const _Float16* bp = bfrag + (size_t)lane * 16;

  for (int nt = 0; nt < NTILES; ++nt) {
    v8f acc = {};
#pragma unroll
    for (int ks = 0; ks < KSTEPS; ++ks) {
      v16h bf = *(const v16h*)(bp + (size_t)(nt * KSTEPS + ks) * 32 * 16);
      acc = __builtin_amdgcn_wmma_f32_16x16x32_f16(
          /*neg_a=*/false, A[ks], /*neg_b=*/false, bf,
          /*c_mod=*/(short)0, acc, /*reuse_a=*/false, /*reuse_b=*/false);
    }

    // ---- fused YOLO decode epilogue for this 16x16 tile ----
    int o = nt * 16 + lm;                            // global out channel
    bool valid = (o < NOUT);
    float bv = valid ? bias[o] : 0.0f;
    int aidx = o / 85;
    int ch   = o - aidx * 85;
    // anchors (w,h) already in pixels (anchor/stride * stride)
    float aw = (aidx == 0) ? 12.0f : ((aidx == 1) ? 19.0f : 40.0f);
    float ah = (aidx == 0) ? 16.0f : ((aidx == 1) ? 36.0f : 28.0f);

#pragma unroll
    for (int r = 0; r < 8; ++r) {
      float v  = acc[r] + bv;
      float sg = __builtin_amdgcn_rcpf(1.0f + __expf(-v));  // sigmoid
      float dv;
      if      (ch == 0) dv = (sg + (float)pcol[r]) * 8.0f;  // bx
      else if (ch == 1) dv = (sg + (float)prow[r]) * 8.0f;  // by
      else if (ch == 2) dv = __expf(v) * aw;                // bw
      else if (ch == 3) dv = __expf(v) * ah;                // bh
      else              dv = sg;                            // conf / cls
      if (valid) {
        int p = pbase + r + 8 * hi;
        size_t idx = ((size_t)(b * 3 + aidx) * NPIX + p) * 85 + ch;
        out[idx] = dv;                                      // each elem once
      }
    }
  }
}

extern "C" void kernel_launch(void* const* d_in, const int* in_sizes, int n_in,
                              void* d_out, int out_size, void* d_ws, size_t ws_size,
                              hipStream_t stream) {
  const float* x    = (const float*)d_in[0];   // [32,256,76,76]
  const float* W    = (const float*)d_in[1];   // [255,256]
  const float* bias = (const float*)d_in[2];   // [255]
  float* out        = (float*)d_out;           // [32, 3*5776, 85]
  _Float16* bfrag   = (_Float16*)d_ws;         // 128 KB swizzled W (f16)

  yolo_prep_b<<<16, 256, 0, stream>>>(W, bfrag);
  const int mtiles = NBATCH * MT_PER_B;        // 11552
  yolo_main<<<mtiles / 8, 256, 0, stream>>>(x, bias, bfrag, out);
}
